// Decoder_83665962926300
// MI455X (gfx1250) — matrix operand order repaired and checked by reading, not compiled, not run on hardware
//
#include <hip/hip_runtime.h>
#include <math.h>

// ---------------------------------------------------------------------------
// Fused 3-layer edge-message-passing decoder for MI455X (gfx1250, wave32).
//  - One workgroup (256 thr / 8 waves) owns NPW=4 nodes (128 edge rows).
//  - Edge features staged once to LDS as bf16; per-layer weights restaged.
//  - Heavy GEMMs (edge MLP layers 1&2) via v_wmma_f32_16x16x32_bf16.
//  - h/node concat columns folded into a per-node "pre" vector (K=512 -> 128).
//  - GEMM3 folded through the k-sum:  sum_k(m2) @ w3^T.
//  - Branch-free gelu (v_tanh_f32 if available, else exp2+rcp sigmoid form).
// ---------------------------------------------------------------------------

typedef unsigned short u16;
typedef unsigned int   u32;

typedef __bf16 v16bf __attribute__((ext_vector_type(16)));
typedef float  v8f   __attribute__((ext_vector_type(8)));

#define NNODES 20000
#define KEDGE  32
#define CCH    128
#define HH     128
#define EINW   512
#define LAY    3
#define NPW    4              // nodes per workgroup
#define ROWS   (NPW * KEDGE)  // 128 edge rows per block
#define SROW   136            // bf16 row stride in LDS (16B-aligned, padded)
#define THREADS 256

static __device__ inline u16 f2bf(float x) {
    u32 u = __builtin_bit_cast(u32, x);
    u32 r = u + 0x7FFFu + ((u >> 16) & 1u);   // round-to-nearest-even
    return (u16)(r >> 16);
}

// Branch-free gelu. Prefer the CDNA5 v_tanh_f32 transcendental; otherwise the
// sigmoid form x*sigmoid(1.702x) with native exp2 + rcp (no EXEC-mask branches).
static __device__ inline float gelu_f(float x) {
#if __has_builtin(__builtin_amdgcn_tanh_f32)
    float t = __builtin_amdgcn_tanh_f32(0.7978845608f * (x + 0.044715f * x * x * x));
    return 0.5f * x * (1.0f + t);
#else
    float e = __builtin_amdgcn_exp2f(x * -2.4554671f);     // exp(-1.702x)
    return x * __builtin_amdgcn_rcpf(1.0f + e);
#endif
}

static __device__ inline float wred(float v) {
#pragma unroll
    for (int m = 16; m >= 1; m >>= 1) v += __shfl_xor(v, m, 32);
    return v;
}

union Frag { v16bf v; uint4 q[2]; };

// A-matrix 16x32 bf16 fragment (CDNA5 layout): lane<16 -> K {0..7,16..23},
// lane>=16 -> K {8..15,24..31}; row = tm + (lane&15). Two 16B LDS loads.
static __device__ inline v16bf load_a(const u16* base, int tm, int kbase, int lane) {
    const int row = tm + (lane & 15);
    const int h   = lane >> 4;
    const u16* p  = base + row * SROW + kbase + h * 8;
    Frag f;
    f.q[0] = *(const uint4*)(p);
    f.q[1] = *(const uint4*)(p + 16);
    return f.v;
}

// B-matrix 32x16 bf16 fragment: lane<16 -> K {0..7,16..23}, lane>=16 -> K {8..15,24..31},
// col n = tn*16 + (lane&15); B[k][n] = W[n][k] (weights stored row-major) so
// two 16B LDS reads from weight row n.
static __device__ inline v16bf load_b(const u16* wbase, int tn, int kbase, int lane) {
    const int n = tn * 16 + (lane & 15);
    const int h = lane >> 4;
    const u16* p = wbase + n * SROW + kbase + h * 8;
    Frag f;
    f.q[0] = *(const uint4*)(p);
    f.q[1] = *(const uint4*)(p + 16);
    return f.v;
}

__global__ __launch_bounds__(THREADS)
void decoder_fused_kernel(const float* __restrict__ node_features,
                          const float* __restrict__ edge_features,
                          const float* __restrict__ mask,
                          const float* __restrict__ w1, const float* __restrict__ b1,
                          const float* __restrict__ w2, const float* __restrict__ b2,
                          const float* __restrict__ w3, const float* __restrict__ b3,
                          const float* __restrict__ n1_s, const float* __restrict__ n1_b,
                          const float* __restrict__ di_w, const float* __restrict__ di_b,
                          const float* __restrict__ do_w, const float* __restrict__ do_b,
                          const float* __restrict__ n2_s, const float* __restrict__ n2_b,
                          float* __restrict__ out) {
    __shared__ u16  sE[ROWS * SROW];   // edge features, bf16 (fixed across layers)
    __shared__ u16  sM[ROWS * SROW];   // gelu(layer1) activations, bf16
    __shared__ u16  sW[HH * SROW];     // staged weight tile, bf16
    __shared__ float sPre[NPW * HH];   // per-node layer-1 pre-activation
    __shared__ float sH[NPW * CCH];    // current h (fp32)
    __shared__ float sNF[NPW * CCH];   // original node features (fixed)
    __shared__ float sS2[NPW * HH];    // sum_k gelu(m2)
    __shared__ float sT[NPW * CCH];    // h after first layernorm
    __shared__ float sG[NPW * HH];     // gelu of node-MLP hidden
    __shared__ float sB2[HH];
    __shared__ float sMask[NPW];

    const int tid   = threadIdx.x;
    const int wv    = tid >> 5;
    const int lane  = tid & 31;
    const int n16   = lane & 15;
    const int hh8   = (lane >> 4) * 8;
    const int node0 = blockIdx.x * NPW;

    // ---- stage edge features (bf16) + node features + mask ----
    {
        const float* eb = edge_features + (size_t)blockIdx.x * (NPW * KEDGE * CCH);
#pragma unroll 4
        for (int q = tid; q < ROWS * CCH / 4; q += THREADS) {
            const int row = q >> 5, c4 = q & 31;
            float4 v = *(const float4*)(eb + (size_t)q * 4);
            u16* dst = &sE[row * SROW + c4 * 4];
            *(u32*)(dst)     = (u32)f2bf(v.x) | ((u32)f2bf(v.y) << 16);
            *(u32*)(dst + 2) = (u32)f2bf(v.z) | ((u32)f2bf(v.w) << 16);
        }
        for (int q = tid; q < NPW * CCH; q += THREADS) {
            float v = node_features[(size_t)node0 * CCH + q];
            sH[q] = v; sNF[q] = v;
        }
        if (tid < NPW) sMask[tid] = mask[node0 + tid];
    }
    __syncthreads();

    for (int l = 0; l < LAY; ++l) {
        // ---- per-node pre-activation: b1 + h@W1a^T + nf@W1b^T (zeros block skipped)
        for (int o = tid; o < NPW * HH; o += THREADS) {
            const int i = o >> 7, hh = o & (HH - 1);
            const float4* wa = (const float4*)(w1 + ((size_t)(l * HH + hh)) * EINW);
            const float4* wb = wa + (CCH / 4);
            float acc = b1[l * HH + hh];
#pragma unroll 8
            for (int q = 0; q < CCH / 4; ++q) {
                float4 a4 = wa[q], b4 = wb[q];
                const int c = q * 4;
                acc += sH[i * CCH + c] * a4.x + sH[i * CCH + c + 1] * a4.y +
                       sH[i * CCH + c + 2] * a4.z + sH[i * CCH + c + 3] * a4.w;
                acc += sNF[i * CCH + c] * b4.x + sNF[i * CCH + c + 1] * b4.y +
                       sNF[i * CCH + c + 2] * b4.z + sNF[i * CCH + c + 3] * b4.w;
            }
            sPre[o] = acc;
        }
        // ---- stage W1c = w1[:, 384:512] as bf16 (+ prefetch next layer's rows)
#pragma unroll 4
        for (int q = tid; q < HH * CCH / 4; q += THREADS) {
            const int hh = q >> 5, c4 = q & 31;
            const float* src = w1 + ((size_t)(l * HH + hh)) * EINW + 384 + c4 * 4;
            float4 v = *(const float4*)src;
            u16* dst = &sW[hh * SROW + c4 * 4];
            *(u32*)(dst)     = (u32)f2bf(v.x) | ((u32)f2bf(v.y) << 16);
            *(u32*)(dst + 2) = (u32)f2bf(v.z) | ((u32)f2bf(v.w) << 16);
            if (l + 1 < LAY) __builtin_prefetch(src + (size_t)HH * EINW, 0, 1);
        }
        __syncthreads();

        // ---- GEMM1: (128 x 128) @ (128 x 128)^T, + pre broadcast, gelu -> sM (bf16)
        {
            const int tm = wv * 16;
            const int nodeL = wv >> 1;
            v16bf af[4];
#pragma unroll
            for (int kk = 0; kk < 4; ++kk) af[kk] = load_a(sE, tm, kk * 32, lane);
#pragma unroll
            for (int tn = 0; tn < 8; ++tn) {
                v8f acc = {};
#pragma unroll
                for (int kk = 0; kk < 4; ++kk) {
                    v16bf bfr = load_b(sW, tn, kk * 32, lane);
                    acc = __builtin_amdgcn_wmma_f32_16x16x32_bf16(
                        false, af[kk], false, bfr, (short)0, acc, false, false);
                }
                const int ncol = tn * 16 + n16;
                const float p = sPre[nodeL * HH + ncol];
#pragma unroll
                for (int r = 0; r < 8; ++r) {
                    float g = gelu_f(acc[r] + p);
                    sM[(tm + r + hh8) * SROW + ncol] = f2bf(g);
                }
            }
        }
        __syncthreads();

        // ---- stage w2 (bf16), b2, zero row-sum accumulator
#pragma unroll 4
        for (int q = tid; q < HH * HH / 4; q += THREADS) {
            const int hh = q >> 5, c4 = q & 31;
            const float* src = w2 + ((size_t)(l * HH + hh)) * HH + c4 * 4;
            float4 v = *(const float4*)src;
            u16* dst = &sW[hh * SROW + c4 * 4];
            *(u32*)(dst)     = (u32)f2bf(v.x) | ((u32)f2bf(v.y) << 16);
            *(u32*)(dst + 2) = (u32)f2bf(v.z) | ((u32)f2bf(v.w) << 16);
            if (l + 1 < LAY) __builtin_prefetch(src + (size_t)HH * HH, 0, 1);
        }
        if (tid < HH) sB2[tid] = b2[l * HH + tid];
        for (int q = tid; q < NPW * HH; q += THREADS) sS2[q] = 0.0f;
        __syncthreads();

        // ---- GEMM2: gelu(m1 @ w2^T + b2), reduced over k into sS2
        {
            const int tm = wv * 16;
            const int nodeL = wv >> 1;
            v16bf af[4];
#pragma unroll
            for (int kk = 0; kk < 4; ++kk) af[kk] = load_a(sM, tm, kk * 32, lane);
#pragma unroll
            for (int tn = 0; tn < 8; ++tn) {
                v8f acc = {};
#pragma unroll
                for (int kk = 0; kk < 4; ++kk) {
                    v16bf bfr = load_b(sW, tn, kk * 32, lane);
                    acc = __builtin_amdgcn_wmma_f32_16x16x32_bf16(
                        false, af[kk], false, bfr, (short)0, acc, false, false);
                }
                const int ncol = tn * 16 + n16;
                const float bb = sB2[ncol];
                float s = 0.0f;
#pragma unroll
                for (int r = 0; r < 8; ++r) s += gelu_f(acc[r] + bb);
                atomicAdd(&sS2[nodeL * HH + ncol], s);   // ds_add_f32
            }
        }
        __syncthreads();

        // ---- h += (sum_k m2)@w3^T/30 (+K*b3/30); layernorm1  (wave i owns node i)
        if (wv < NPW) {
            const int i = wv;
            float u[4];
            float ls = 0.0f, lq = 0.0f;
#pragma unroll
            for (int j = 0; j < 4; ++j) {
                const int c = lane * 4 + j;
                const float4* wr = (const float4*)(w3 + ((size_t)(l * CCH + c)) * HH);
                float acc = 0.0f;
#pragma unroll 8
                for (int q = 0; q < HH / 4; ++q) {
                    float4 v = wr[q];
                    const int hb = q * 4;
                    acc += sS2[i * HH + hb] * v.x + sS2[i * HH + hb + 1] * v.y +
                           sS2[i * HH + hb + 2] * v.z + sS2[i * HH + hb + 3] * v.w;
                }
                float uu = sH[i * CCH + c] +
                           (acc + (float)KEDGE * b3[l * CCH + c]) * (1.0f / 30.0f);
                u[j] = uu; ls += uu; lq += uu * uu;
            }
            ls = wred(ls); lq = wred(lq);
            const float mean = ls * (1.0f / 128.0f);
            const float inv  = __builtin_amdgcn_rsqf(lq * (1.0f / 128.0f) - mean * mean + 1e-5f);
#pragma unroll
            for (int j = 0; j < 4; ++j) {
                const int c = lane * 4 + j;
                sT[i * CCH + c] = (u[j] - mean) * inv * n1_s[l * CCH + c] + n1_b[l * CCH + c];
            }
        }
        __syncthreads();

        // ---- node MLP hidden: g = gelu(h1 @ di_w^T + di_b)
        if (wv < NPW) {
            const int i = wv;
#pragma unroll
            for (int j = 0; j < 4; ++j) {
                const int hh = lane * 4 + j;
                const float4* wr = (const float4*)(di_w + ((size_t)(l * HH + hh)) * CCH);
                float acc = di_b[l * HH + hh];
#pragma unroll 8
                for (int q = 0; q < CCH / 4; ++q) {
                    float4 v = wr[q];
                    const int cb = q * 4;
                    acc += sT[i * CCH + cb] * v.x + sT[i * CCH + cb + 1] * v.y +
                           sT[i * CCH + cb + 2] * v.z + sT[i * CCH + cb + 3] * v.w;
                }
                sG[i * HH + hh] = gelu_f(acc);
            }
        }
        __syncthreads();

        // ---- d = g @ do_w^T + do_b; layernorm2(h1 + d); mask; write back to sH
        if (wv < NPW) {
            const int i = wv;
            float u2[4];
            float ls = 0.0f, lq = 0.0f;
#pragma unroll
            for (int j = 0; j < 4; ++j) {
                const int c = lane * 4 + j;
                const float4* wr = (const float4*)(do_w + ((size_t)(l * CCH + c)) * HH);
                float acc = do_b[l * CCH + c];
#pragma unroll 8
                for (int q = 0; q < HH / 4; ++q) {
                    float4 v = wr[q];
                    const int hb = q * 4;
                    acc += sG[i * HH + hb] * v.x + sG[i * HH + hb + 1] * v.y +
                           sG[i * HH + hb + 2] * v.z + sG[i * HH + hb + 3] * v.w;
                }
                float uu = sT[i * CCH + c] + acc;
                u2[j] = uu; ls += uu; lq += uu * uu;
            }
            ls = wred(ls); lq = wred(lq);
            const float mean = ls * (1.0f / 128.0f);
            const float inv  = __builtin_amdgcn_rsqf(lq * (1.0f / 128.0f) - mean * mean + 1e-5f);
            const float mk   = sMask[i];
#pragma unroll
            for (int j = 0; j < 4; ++j) {
                const int c = lane * 4 + j;
                float h2 = (u2[j] - mean) * inv * n2_s[l * CCH + c] + n2_b[l * CCH + c];
                sH[i * CCH + c] = mk * h2;
            }
        }
        __syncthreads();
    }

    // ---- write result
    for (int q = tid; q < NPW * CCH; q += THREADS)
        out[(size_t)node0 * CCH + q] = sH[q];
}

extern "C" void kernel_launch(void* const* d_in, const int* in_sizes, int n_in,
                              void* d_out, int out_size, void* d_ws, size_t ws_size,
                              hipStream_t stream) {
    (void)in_sizes; (void)n_in; (void)out_size; (void)d_ws; (void)ws_size;
    const float* node_features = (const float*)d_in[0];
    const float* edge_features = (const float*)d_in[1];
    const float* mask          = (const float*)d_in[2];
    const float* w1   = (const float*)d_in[3];
    const float* b1   = (const float*)d_in[4];
    const float* w2   = (const float*)d_in[5];
    const float* b2   = (const float*)d_in[6];
    const float* w3   = (const float*)d_in[7];
    const float* b3   = (const float*)d_in[8];
    const float* n1_s = (const float*)d_in[9];
    const float* n1_b = (const float*)d_in[10];
    const float* di_w = (const float*)d_in[11];
    const float* di_b = (const float*)d_in[12];
    const float* do_w = (const float*)d_in[13];
    const float* do_b = (const float*)d_in[14];
    const float* n2_s = (const float*)d_in[15];
    const float* n2_b = (const float*)d_in[16];

    dim3 grid(NNODES / NPW), block(THREADS);
    decoder_fused_kernel<<<grid, block, 0, stream>>>(
        node_features, edge_features, mask,
        w1, b1, w2, b2, w3, b3,
        n1_s, n1_b, di_w, di_b, do_w, do_b, n2_s, n2_b,
        (float*)d_out);
}
